// RNNModel_37598143710010
// MI455X (gfx1250) — compile-verified, matrix operand order checked by reading, not run on hardware
//
#include <hip/hip_runtime.h>
#include <hip/hip_bf16.h>

#define B_  2048
#define T_  100
#define U_  512
#define C_  10

typedef __attribute__((ext_vector_type(16))) __bf16 v16bf;
typedef __attribute__((ext_vector_type(8)))  float  v8f;

union FragU { uint4 u[2]; v16bf v; };

__device__ __forceinline__ unsigned short f2bf(float f) {
    unsigned int u = __float_as_uint(f);
    unsigned int r = (u + 0x7fffu + ((u >> 16) & 1u)) >> 16;
    return (unsigned short)r;
}
__device__ __forceinline__ float bf2f(unsigned short h) {
    return __uint_as_float(((unsigned int)h) << 16);
}
__device__ __forceinline__ float sigmoid_f(float v) {
    return 1.0f / (1.0f + __expf(-v));
}
__device__ __forceinline__ float tanh_f(float v) {
    float a = fabsf(v);
    float e = __expf(-2.0f * a);
    float r = (1.0f - e) / (1.0f + e);
    return copysignf(r, v);
}

// Async DMA: global memory -> LDS, 16 bytes per lane, no data VGPRs.
// Tracked by ASYNCcnt (CDNA5 Tensor/Async path).
__device__ __forceinline__ void async_b128(unsigned lds_off, const void* gptr) {
    unsigned long long ga = (unsigned long long)(uintptr_t)gptr;
    asm volatile("global_load_async_to_lds_b128 %0, %1, off"
                 :: "v"(lds_off), "v"(ga) : "memory");
}
__device__ __forceinline__ void wait_async0() {
    asm volatile("s_wait_asynccnt 0x0" ::: "memory");
}

// Build gate-major transposed bf16 weights: WT[R=4U rows][Ktot cols]
__global__ __launch_bounds__(256) void convert_weights(
    const float* __restrict__ W, const float* __restrict__ Uw,
    unsigned short* __restrict__ WT, int Ktot)
{
    int total = 4 * U_ * Ktot;
    int i = blockIdx.x * 256 + threadIdx.x;
    if (i >= total) return;
    int R = i / Ktot;
    int k = i - R * Ktot;
    float v;
    if (Ktot == 1024) v = (k < 512) ? W[(size_t)k * (4 * U_) + R]
                                    : Uw[(size_t)(k - 512) * (4 * U_) + R];
    else              v = Uw[(size_t)k * (4 * U_) + R];
    WT[(size_t)i] = f2bf(v);
}

// Fused LSTM step. Block: 128 batch rows x 32 hidden units x 4 gate strips,
// 4 waves; each wave owns 32 rows -> 16 x v_wmma_f32_16x16x32_bf16 per
// 32-wide K step, each B fragment reused for two M tiles.
// Double-buffered LDS filled by async global->LDS DMA (no staging VGPRs):
// issue DMA for tile k+1, compute tile k, s_wait_asynccnt + one barrier.
__global__ __launch_bounds__(128) void lstm_step_kernel(
    const unsigned short* __restrict__ A0,   // phase-0 activations (B x 512) bf16
    const unsigned short* __restrict__ A1,   // phase-1 activations or null
    const unsigned short* __restrict__ WT,   // (4U x Ktot) bf16, gate-major rows
    int Ktot,
    const float* __restrict__ bias,          // 4U
    const float* __restrict__ W0row,         // 1x4U fp32 (layer0) or null
    const float* __restrict__ x,             // (B,T) fp32 (layer0) or null
    const int*   __restrict__ mask,          // (B,T)
    const unsigned short* __restrict__ h_in, // (B x U) bf16 (old h, mask carry)
    unsigned short* __restrict__ h_out,      // (B x U) bf16
    float* __restrict__ cstate,              // (B x U) fp32, in place
    int t)
{
    constexpr int ST = 40; // padded LDS stride (halfwords): conflict-free b128 frags
    __shared__ __align__(16) unsigned short As[2][128 * ST];
    __shared__ __align__(16) unsigned short Bs[2][128 * ST];
    __shared__ float xs[128];
    __shared__ int   ms[128];

    const int tid = threadIdx.x;
    const int b0  = blockIdx.x * 128;  // batch rows
    const int n0  = blockIdx.y * 32;   // hidden units

    ms[tid] = mask[(size_t)(b0 + tid) * T_ + t];
    xs[tid] = x ? x[(size_t)(b0 + tid) * T_ + t] : 0.0f;

    const int ntiles = Ktot >> 5;      // 16 or 32 K-tiles of width 32

    // LDS byte offsets (low 32 bits of shared-aperture flat address == LDS offset)
    const unsigned asA = (unsigned)(uintptr_t)(&As[0][0]);
    const unsigned asB = (unsigned)(uintptr_t)(&Bs[0][0]);
    constexpr unsigned BUFB = 128u * ST * 2u;   // bytes per buffer

    auto issue_async = [&](int it, int buf) {
        const unsigned short* srcA = (it < 16) ? A0 : A1;
        const int kk   = (it & 15) * 32;
        const int kofs = (it < 16) ? 0 : 512;
#pragma unroll
        for (int i = 0; i < 4; ++i) {
            int li = tid + i * 128;
            int row = li >> 2, seg = li & 3;
            unsigned loff = (unsigned)(row * ST + seg * 8) * 2u + buf * BUFB;
            async_b128(asA + loff, srcA + (size_t)(b0 + row) * U_ + kk + seg * 8);
            size_t Rw = (size_t)((row >> 5) * U_ + n0 + (row & 31));
            async_b128(asB + loff, WT + Rw * Ktot + kofs + kk + seg * 8);
        }
    };

    issue_async(0, 0);
    wait_async0();
    __syncthreads();

    v8f acc[2][4][2] = {};             // [m][gate][j]

    const int wv   = tid >> 5;
    const int lane = tid & 31;
    const int lrow = lane & 15;
    const int kh   = (lane >> 4) << 3; // 0 or 8 (K half select per ISA layout)

    for (int it = 0; it < ntiles; ++it) {
        const int cur = it & 1;
        if (it + 1 < ntiles) issue_async(it + 1, cur ^ 1); // DMA overlaps wmma

        FragU fa[2];
#pragma unroll
        for (int m = 0; m < 2; ++m) {
            const unsigned short* ap = As[cur] + (wv * 32 + m * 16 + lrow) * ST + kh;
            fa[m].u[0] = *(const uint4*)ap;
            fa[m].u[1] = *(const uint4*)(ap + 16);
        }
#pragma unroll
        for (int g = 0; g < 4; ++g) {
#pragma unroll
            for (int j = 0; j < 2; ++j) {
                FragU fb;
                const unsigned short* bp = Bs[cur] + (g * 32 + j * 16 + lrow) * ST + kh;
                fb.u[0] = *(const uint4*)bp;
                fb.u[1] = *(const uint4*)(bp + 16);
#pragma unroll
                for (int m = 0; m < 2; ++m)
                    acc[m][g][j] = __builtin_amdgcn_wmma_f32_16x16x32_bf16(
                        false, fa[m].v, false, fb.v, (short)0, acc[m][g][j], false, false);
            }
        }
        if (it + 1 < ntiles) wait_async0();  // my DMA for next buffer done
        __syncthreads();                     // everyone's DMA visible
    }

    // Epilogue: C/D layout -> row = wv*32 + m*16 + v + (lane>=16?8:0),
    //                         col = j*16 + lane%16
    const int lcol   = lane & 15;
    const int rshift = (lane >> 4) * 8;
#pragma unroll
    for (int j = 0; j < 2; ++j) {
        const int n = n0 + j * 16 + lcol;
        const float bi = bias[n], bff = bias[U_ + n], bg = bias[2 * U_ + n], bo = bias[3 * U_ + n];
        float wi = 0.f, wf = 0.f, wg = 0.f, wo = 0.f;
        if (W0row) { wi = W0row[n]; wf = W0row[U_ + n]; wg = W0row[2 * U_ + n]; wo = W0row[3 * U_ + n]; }
#pragma unroll
        for (int m = 0; m < 2; ++m) {
#pragma unroll
            for (int v = 0; v < 8; ++v) {
                const int rl = wv * 32 + m * 16 + v + rshift;
                const size_t idx = (size_t)(b0 + rl) * U_ + n;
                float zi = acc[m][0][j][v] + bi;
                float zf = acc[m][1][j][v] + bff;
                float zg = acc[m][2][j][v] + bg;
                float zo = acc[m][3][j][v] + bo;
                if (W0row) {
                    float xv = xs[rl];
                    zi += xv * wi; zf += xv * wf; zg += xv * wg; zo += xv * wo;
                }
                float ig = sigmoid_f(zi);
                float fg = sigmoid_f(zf);
                float gg = tanh_f(zg);
                float og = sigmoid_f(zo);
                float co = cstate[idx];
                float cn = fg * co + ig * gg;
                float hn = og * tanh_f(cn);
                if (!ms[rl]) { cn = co; hn = bf2f(h_in[idx]); }
                cstate[idx] = cn;
                h_out[idx]  = f2bf(hn);
            }
        }
    }
}

// logits = h_top @ Wd + bd; softmax. One wave per batch row (trivial FLOPs).
__global__ __launch_bounds__(32) void dense_softmax_kernel(
    const unsigned short* __restrict__ h, const float* __restrict__ Wd,
    const float* __restrict__ bd, float* __restrict__ out)
{
    const int row = blockIdx.x;
    const int lane = threadIdx.x;
    float acc[C_];
#pragma unroll
    for (int c = 0; c < C_; ++c) acc[c] = 0.f;
    for (int k = lane; k < U_; k += 32) {
        float hv = bf2f(h[(size_t)row * U_ + k]);
#pragma unroll
        for (int c = 0; c < C_; ++c) acc[c] += hv * Wd[(size_t)k * C_ + c];
    }
#pragma unroll
    for (int c = 0; c < C_; ++c)
        for (int off = 16; off > 0; off >>= 1)
            acc[c] += __shfl_xor(acc[c], off, 32);
    if (lane == 0) {
        float lg[C_], m = -1e30f;
#pragma unroll
        for (int c = 0; c < C_; ++c) { lg[c] = acc[c] + bd[c]; m = fmaxf(m, lg[c]); }
        float s = 0.f;
#pragma unroll
        for (int c = 0; c < C_; ++c) { lg[c] = __expf(lg[c] - m); s += lg[c]; }
        float inv = 1.0f / s;
#pragma unroll
        for (int c = 0; c < C_; ++c) out[(size_t)row * C_ + c] = lg[c] * inv;
    }
}

extern "C" void kernel_launch(void* const* d_in, const int* in_sizes, int n_in,
                              void* d_out, int out_size, void* d_ws, size_t ws_size,
                              hipStream_t stream) {
    const float* x    = (const float*)d_in[0];
    const int*   mask = (const int*)  d_in[1];
    const float* W0   = (const float*)d_in[2];
    const float* U0   = (const float*)d_in[3];
    const float* b0   = (const float*)d_in[4];
    const float* W1   = (const float*)d_in[5];
    const float* U1   = (const float*)d_in[6];
    const float* b1   = (const float*)d_in[7];
    const float* W2   = (const float*)d_in[8];
    const float* U2   = (const float*)d_in[9];
    const float* b2   = (const float*)d_in[10];
    const float* Wd   = (const float*)d_in[11];
    const float* bd   = (const float*)d_in[12];

    char* ws = (char*)d_ws;
    size_t off = 0;
    const size_t WT0_sz  = (size_t)4 * U_ * 512  * sizeof(unsigned short);
    const size_t WT12_sz = (size_t)4 * U_ * 1024 * sizeof(unsigned short);
    const size_t H_sz    = (size_t)B_ * U_ * sizeof(unsigned short);
    const size_t Cst_sz  = (size_t)B_ * U_ * sizeof(float);

    unsigned short* WT0 = (unsigned short*)(ws + off); off += WT0_sz;
    unsigned short* WT1 = (unsigned short*)(ws + off); off += WT12_sz;
    unsigned short* WT2 = (unsigned short*)(ws + off); off += WT12_sz;
    char* zero_base = ws + off;
    unsigned short* h[3][2];
    for (int l = 0; l < 3; ++l)
        for (int p = 0; p < 2; ++p) { h[l][p] = (unsigned short*)(ws + off); off += H_sz; }
    float* cst[3];
    for (int l = 0; l < 3; ++l) { cst[l] = (float*)(ws + off); off += Cst_sz; }
    size_t zero_bytes = (size_t)((ws + off) - zero_base);

    // Zero h (bf16 zero == 0x0000) and c states. Capturable memset node.
    hipMemsetAsync(zero_base, 0, zero_bytes, stream);

    // Weight conversion / transpose (runs every call; deterministic, cheap vs GEMMs).
    {
        int tot0 = 4 * U_ * 512, tot1 = 4 * U_ * 1024;
        convert_weights<<<(tot0 + 255) / 256, 256, 0, stream>>>(nullptr, U0, WT0, 512);
        convert_weights<<<(tot1 + 255) / 256, 256, 0, stream>>>(W1, U1, WT1, 1024);
        convert_weights<<<(tot1 + 255) / 256, 256, 0, stream>>>(W2, U2, WT2, 1024);
    }

    dim3 grid(B_ / 128, U_ / 32);
    for (int t = 0; t < T_; ++t) {
        const int pi = t & 1, po = 1 - pi;
        // layer 0: A = h0_prev (K=512) + rank-1 x*W0 term in epilogue
        lstm_step_kernel<<<grid, 128, 0, stream>>>(
            h[0][pi], nullptr, WT0, 512, b0, W0, x, mask,
            h[0][pi], h[0][po], cst[0], t);
        // layer 1: A = [h0_new | h1_prev], K=1024
        lstm_step_kernel<<<grid, 128, 0, stream>>>(
            h[0][po], h[1][pi], WT1, 1024, b1, nullptr, nullptr, mask,
            h[1][pi], h[1][po], cst[1], t);
        // layer 2: A = [h1_new | h2_prev], K=1024
        lstm_step_kernel<<<grid, 128, 0, stream>>>(
            h[1][po], h[2][pi], WT2, 1024, b2, nullptr, nullptr, mask,
            h[2][pi], h[2][po], cst[2], t);
    }

    dense_softmax_kernel<<<B_, 32, 0, stream>>>(h[2][T_ & 1], Wd, bd, (float*)d_out);
}
